// Multi_query_Attention_58686433133027
// MI455X (gfx1250) — matrix operand order repaired and checked by reading, not compiled, not run on hardware
//
#include <hip/hip_runtime.h>

typedef __attribute__((ext_vector_type(16))) __bf16 v16bf;
typedef __attribute__((ext_vector_type(8)))  __bf16 v8bf;
typedef __attribute__((ext_vector_type(4)))  __bf16 v4bf;
typedef __attribute__((ext_vector_type(8)))  float  v8f;
typedef __attribute__((ext_vector_type(4)))  int    v4i;

#define SEQ  4096
#define EMB  1024
#define NH   16
#define HDIM 64

// Async global->LDS copy (ASYNCcnt-tracked), if the toolchain exposes it.
#if defined(__has_builtin)
#  if __has_builtin(__builtin_amdgcn_global_load_async_to_lds_b128) && \
      __has_builtin(__builtin_amdgcn_s_wait_asynccnt)
#    define HAVE_ASYNC_LDS 1
#  endif
#endif
#ifndef HAVE_ASYNC_LDS
#  define HAVE_ASYNC_LDS 0
#endif

#if HAVE_ASYNC_LDS
typedef __attribute__((address_space(1))) v4i gv4i;
typedef __attribute__((address_space(3))) v4i lv4i;
static __device__ __forceinline__ void async_copy16(const void* g, void* l) {
  __builtin_amdgcn_global_load_async_to_lds_b128((gv4i*)(v4i*)g, (lv4i*)(v4i*)l, 0, 0);
}
#endif

static __device__ __forceinline__ __bf16 f2bf(float f) {
  union { float f; unsigned u; } a; a.f = f;
  unsigned r = a.u + 0x7fffu + ((a.u >> 16) & 1u);
  unsigned short h = (unsigned short)(r >> 16);
  __bf16 o; __builtin_memcpy(&o, &h, 2);
  return o;
}

#define WMMA_BF16(A, B, C) \
  __builtin_amdgcn_wmma_f32_16x16x32_bf16(false, (A), false, (B), (short)0, (C), false, false)

union bfrag { v16bf v; v8bf h2[2]; };

// ---------------------------------------------------------------------------
// fp32 -> bf16 convert (vectorized x4)
// ---------------------------------------------------------------------------
__global__ __launch_bounds__(256) void cvt_bf16_kernel(const float* __restrict__ in,
                                                       __bf16* __restrict__ out, int n4) {
  int i = blockIdx.x * 256 + threadIdx.x;
  if (i < n4) {
    float4 f = ((const float4*)in)[i];
    v4bf o;
    o[0] = f2bf(f.x); o[1] = f2bf(f.y); o[2] = f2bf(f.z); o[3] = f2bf(f.w);
    ((v4bf*)out)[i] = o;
  }
}

// ---------------------------------------------------------------------------
// Tiled WMMA GEMM: C[M,N] = A[M,K] @ B[K,N]  (A,B bf16 row-major)
// Block tile 64 x (NT*32), 8 waves (4 in M x 2 in N); each wave computes
// 16 x (NT*16) with NT WMMAs per 32-deep K-step (A fragment reused NT times).
// LDS: A row-major (pitch 40), B transposed [n][k] (pitch 40) so both
// fragment loads are contiguous, 16B-aligned ds_load_b128.
// F32OUT: fp32 output + bias; else bf16 output.
// ---------------------------------------------------------------------------
template <bool F32OUT, int NT>
__global__ __launch_bounds__(256) void gemm_wmma_kernel(
    const __bf16* __restrict__ A, const __bf16* __restrict__ B,
    void* __restrict__ Cout, const float* __restrict__ bias,
    int M, int N, int K) {
  constexpr int BN = NT * 32;            // block tile N
  __shared__ __bf16 ldsA[64 * 40];       // [row][k]
  __shared__ __bf16 ldsBt[BN * 40];      // [n][k]

  const int tid  = threadIdx.x;
  const int wave = tid >> 5;
  const int lane = tid & 31;
  const int lh   = lane & 15;
  const int sel8  = (lane < 16) ? 0 : 8;
  const int sel16 = (lane < 16) ? 0 : 16;

  const int m0 = blockIdx.y * 64;
  const int n0 = blockIdx.x * BN;
  const int wm = (wave & 3) * 16;          // wave row offset in tile
  const int wn = (wave >> 2) * (NT * 16);  // wave col offset in tile

  v8f acc[NT];
#pragma unroll
  for (int t = 0; t < NT; ++t) acc[t] = v8f{};

  for (int kk = 0; kk < K; kk += 32) {
    __syncthreads();
    {  // stage A tile 64x32 (contiguous 16B per lane)
      int row = tid >> 2;
      int c   = (tid & 3) * 8;
      const __bf16* g = A + (size_t)(m0 + row) * K + kk + c;
      __bf16* l = &ldsA[row * 40 + c];
#if HAVE_ASYNC_LDS
      async_copy16(g, l);
#else
      *(v8bf*)l = *(const v8bf*)g;
#endif
    }
    {  // stage B tile 32xBN transposed -> ldsBt[n][k]
#pragma unroll
      for (int idx = tid * 8; idx < 32 * BN; idx += 256 * 8) {
        int krow = idx / BN;
        int c    = idx % BN;
        const __bf16* g = B + (size_t)(kk + krow) * N + n0 + c;
        if (kk + 32 < K)
          __builtin_prefetch((const void*)(g + (size_t)32 * N), 0, 1);
        v8bf vb = *(const v8bf*)g;
#pragma unroll
        for (int j = 0; j < 8; ++j) ldsBt[(c + j) * 40 + krow] = vb[j];
      }
    }
#if HAVE_ASYNC_LDS
    __builtin_amdgcn_s_wait_asynccnt(0);
#endif
    __syncthreads();

    bfrag af;
    {
      const __bf16* p = &ldsA[(wm + lh) * 40 + sel8];
      af.h2[0] = *(const v8bf*)(p);
      af.h2[1] = *(const v8bf*)(p + 16);
    }
#pragma unroll
    for (int t = 0; t < NT; ++t) {
      bfrag bf;
      const __bf16* p = &ldsBt[(wn + t * 16 + lh) * 40 + sel8];
      bf.h2[0] = *(const v8bf*)(p);
      bf.h2[1] = *(const v8bf*)(p + 16);
      acc[t] = WMMA_BF16(af.v, bf.v, acc[t]);
    }
  }

  const int rbase = m0 + wm + ((lane < 16) ? 0 : 8);
  if (F32OUT) {
    float* C = (float*)Cout;
#pragma unroll
    for (int t = 0; t < NT; ++t) {
      int c0 = n0 + wn + t * 16 + lh;
      float bz = bias ? bias[c0] : 0.f;
#pragma unroll
      for (int e = 0; e < 8; ++e)
        C[(size_t)(rbase + e) * N + c0] = acc[t][e] + bz;
    }
  } else {
    __bf16* C = (__bf16*)Cout;
#pragma unroll
    for (int t = 0; t < NT; ++t) {
      int c0 = n0 + wn + t * 16 + lh;
#pragma unroll
      for (int e = 0; e < 8; ++e)
        C[(size_t)(rbase + e) * N + c0] = f2bf(acc[t][e]);
    }
  }
}

// ---------------------------------------------------------------------------
// Flash-style causal MQA attention.
// Q:[S,E] (head h at cols h*64), K:[S,64], V:[S,64], Og:[S,E] (bf16).
// grid = (S/128, H). Block 256 thr = 8 waves; wave w owns queries
// [qtile*128 + w*16, +16). Key loop in blocks of 32, online softmax.
// ---------------------------------------------------------------------------
__global__ __launch_bounds__(256) void mqa_attn_kernel(
    const __bf16* __restrict__ Q, const __bf16* __restrict__ Kg,
    const __bf16* __restrict__ Vg, __bf16* __restrict__ Og) {
  __shared__ __bf16 ldsVt[HDIM * 40];   // [hd][key], pitch 40
  __shared__ __bf16 ldsP[8][16 * 40];   // per-wave P tile [row][key], pitch 40

  const int tid  = threadIdx.x;
  const int wave = tid >> 5;
  const int lane = tid & 31;
  const int lh   = lane & 15;
  const int sel8  = (lane < 16) ? 0 : 8;
  const int sel16 = (lane < 16) ? 0 : 16;
  const int rowoff = (lane < 16) ? 0 : 8;

  const int h     = blockIdx.y;
  const int qtile = blockIdx.x;
  const int qbase = qtile * 128 + wave * 16;
  const int qrow  = qbase + lh;

  // Q fragments for the two 32-wide hd chunks (A-matrix layout, loaded once)
  bfrag qf[2];
#pragma unroll
  for (int s = 0; s < 2; ++s) {
    const __bf16* p = Q + (size_t)qrow * EMB + h * HDIM + s * 32 + sel8;
    qf[s].h2[0] = *(const v8bf*)(p);
    qf[s].h2[1] = *(const v8bf*)(p + 16);
  }

  v8f oacc[4] = {{}, {}, {}, {}};
  float mrow[8], lrow[8];
#pragma unroll
  for (int e = 0; e < 8; ++e) { mrow[e] = -1e30f; lrow[e] = 0.f; }

  const int kend = qtile * 128 + 128;
  for (int kb = 0; kb < kend; kb += 32) {
    __syncthreads();
    {  // stage V tile [32 keys][64 hd] transposed -> ldsVt[hd][key]
      int key = tid >> 3;
      int c   = (tid & 7) * 8;
      const __bf16* g = Vg + (size_t)(kb + key) * HDIM + c;
      if (kb + 32 < kend) __builtin_prefetch((const void*)(g + 32 * HDIM), 0, 1);
      v8bf vv = *(const v8bf*)g;
#pragma unroll
      for (int j = 0; j < 8; ++j) ldsVt[(c + j) * 40 + key] = vv[j];
    }
    __syncthreads();

    // scores S = Q @ K^T  (two key n-tiles of 16, two hd K-steps of 32)
    v8f s0 = {}; v8f s1 = {};
#pragma unroll
    for (int st = 0; st < 2; ++st) {
      bfrag b0, b1;
      const __bf16* p0 = Kg + (size_t)(kb + lh) * HDIM + st * 32 + sel8;
      b0.h2[0] = *(const v8bf*)(p0);
      b0.h2[1] = *(const v8bf*)(p0 + 16);
      const __bf16* p1 = Kg + (size_t)(kb + 16 + lh) * HDIM + st * 32 + sel8;
      b1.h2[0] = *(const v8bf*)(p1);
      b1.h2[1] = *(const v8bf*)(p1 + 16);
      s0 = WMMA_BF16(qf[st].v, b0.v, s0);
      s1 = WMMA_BF16(qf[st].v, b1.v, s1);
    }

    // scale + causal mask (C layout: lane = key col, element = query row)
    const int key0 = kb + lh;
    const int key1 = kb + 16 + lh;
    float corr[8];
#pragma unroll
    for (int e = 0; e < 8; ++e) {
      int r = qbase + rowoff + e;
      float a0 = s0[e] * 0.125f;
      float a1 = s1[e] * 0.125f;
      if (key0 > r) a0 = -3e38f;
      if (key1 > r) a1 = -3e38f;
      // row max over the 16 lanes of this half-wave
      float mx = fmaxf(a0, a1);
      mx = fmaxf(mx, __shfl_xor(mx, 1));
      mx = fmaxf(mx, __shfl_xor(mx, 2));
      mx = fmaxf(mx, __shfl_xor(mx, 4));
      mx = fmaxf(mx, __shfl_xor(mx, 8));
      float mnew = fmaxf(mrow[e], mx);
      float p0 = __expf(a0 - mnew);
      float p1 = __expf(a1 - mnew);
      float rs = p0 + p1;
      rs += __shfl_xor(rs, 1);
      rs += __shfl_xor(rs, 2);
      rs += __shfl_xor(rs, 4);
      rs += __shfl_xor(rs, 8);
      corr[e] = __expf(mrow[e] - mnew);
      mrow[e] = mnew;
      lrow[e] = lrow[e] * corr[e] + rs;
      // spill P (C layout) to per-wave LDS tile, row-major [16][32 keys]
      ldsP[wave][(rowoff + e) * 40 + lh]      = f2bf(p0);
      ldsP[wave][(rowoff + e) * 40 + 16 + lh] = f2bf(p1);
    }

    // rescale running O
#pragma unroll
    for (int t = 0; t < 4; ++t)
#pragma unroll
      for (int e = 0; e < 8; ++e) oacc[t][e] *= corr[e];

    // reload P in A-matrix layout (wave-private LDS, DS ops are in-order)
    bfrag pf;
    {
      const __bf16* p = &ldsP[wave][lh * 40 + sel8];
      pf.h2[0] = *(const v8bf*)(p);
      pf.h2[1] = *(const v8bf*)(p + 16);
    }

    // O += P @ V  (four hd n-tiles of 16)
#pragma unroll
    for (int t = 0; t < 4; ++t) {
      bfrag vf;
      const __bf16* p = &ldsVt[(t * 16 + lh) * 40 + sel8];
      vf.h2[0] = *(const v8bf*)(p);
      vf.h2[1] = *(const v8bf*)(p + 16);
      oacc[t] = WMMA_BF16(pf.v, vf.v, oacc[t]);
    }
  }

  // epilogue: O / l -> bf16 into Og[qrow][h*64 + col]
#pragma unroll
  for (int e = 0; e < 8; ++e) {
    int r = qbase + rowoff + e;
    float inv = 1.0f / lrow[e];
    size_t base = (size_t)r * EMB + h * HDIM;
#pragma unroll
    for (int t = 0; t < 4; ++t) Og[base + t * 16 + lh] = f2bf(oacc[t][e] * inv);
  }
}

// ---------------------------------------------------------------------------
// Host launcher
// ---------------------------------------------------------------------------
extern "C" void kernel_launch(void* const* d_in, const int* in_sizes, int n_in,
                              void* d_out, int out_size, void* d_ws, size_t ws_size,
                              hipStream_t stream) {
  (void)in_sizes; (void)n_in; (void)out_size; (void)ws_size;
  const float* x  = (const float*)d_in[0];
  const float* Wq = (const float*)d_in[1];
  const float* Wk = (const float*)d_in[2];
  const float* Wv = (const float*)d_in[3];
  const float* Wo = (const float*)d_in[4];
  const float* bo = (const float*)d_in[5];
  float* out = (float*)d_out;

  char* ws = (char*)d_ws;
  size_t off = 0;
  __bf16* xb   = (__bf16*)(ws + off); off += (size_t)SEQ * EMB * 2;
  __bf16* Wqb  = (__bf16*)(ws + off); off += (size_t)EMB * EMB * 2;
  __bf16* Wkb  = (__bf16*)(ws + off); off += (size_t)EMB * HDIM * 2;
  __bf16* Wvb  = (__bf16*)(ws + off); off += (size_t)EMB * HDIM * 2;
  __bf16* Wob  = (__bf16*)(ws + off); off += (size_t)EMB * EMB * 2;
  __bf16* qb   = (__bf16*)(ws + off); off += (size_t)SEQ * EMB * 2;
  __bf16* kb   = (__bf16*)(ws + off); off += (size_t)SEQ * HDIM * 2;
  __bf16* vb   = (__bf16*)(ws + off); off += (size_t)SEQ * HDIM * 2;
  __bf16* attb = (__bf16*)(ws + off); off += (size_t)SEQ * EMB * 2;

  auto cvt = [&](const float* src, __bf16* dst, int n) {
    int n4 = n / 4;
    cvt_bf16_kernel<<<(n4 + 255) / 256, 256, 0, stream>>>(src, dst, n4);
  };
  cvt(x,  xb,  SEQ * EMB);
  cvt(Wq, Wqb, EMB * EMB);
  cvt(Wk, Wkb, EMB * HDIM);
  cvt(Wv, Wvb, EMB * HDIM);
  cvt(Wo, Wob, EMB * EMB);

  // q = x @ Wq   [4096,1024]  (block tile 64x128)
  gemm_wmma_kernel<false, 4><<<dim3(EMB / 128, SEQ / 64), 256, 0, stream>>>(
      xb, Wqb, (void*)qb, nullptr, SEQ, EMB, EMB);
  // k = x @ Wk   [4096,64]   (block tile 64x64)
  gemm_wmma_kernel<false, 2><<<dim3(HDIM / 64, SEQ / 64), 256, 0, stream>>>(
      xb, Wkb, (void*)kb, nullptr, SEQ, HDIM, EMB);
  // v = x @ Wv   [4096,64]
  gemm_wmma_kernel<false, 2><<<dim3(HDIM / 64, SEQ / 64), 256, 0, stream>>>(
      xb, Wvb, (void*)vb, nullptr, SEQ, HDIM, EMB);

  // causal MQA flash attention -> attb [4096,1024] bf16
  mqa_attn_kernel<<<dim3(SEQ / 128, NH), 256, 0, stream>>>(qb, kb, vb, attb);

  // out = attb @ Wo + bo  (fp32, block tile 64x128)
  gemm_wmma_kernel<true, 4><<<dim3(EMB / 128, SEQ / 64), 256, 0, stream>>>(
      attb, Wob, (void*)out, bo, SEQ, EMB, EMB);
}
